// GHM_Loss_40175124087465
// MI455X (gfx1250) — matrix operand order repaired and checked by reading, not compiled, not run on hardware
//
#include <hip/hip_runtime.h>

typedef __attribute__((ext_vector_type(4)))  float     f4;
typedef __attribute__((ext_vector_type(8)))  float     v8f;
typedef __attribute__((ext_vector_type(16))) _Float16  v16h;

#define GHM_BINS 10
#define GHM_MOM  0.75f

// ws layout (48 floats): [0..15]=count per bin, [16..31]=ce_hi sum, [32..47]=ce_lo sum
__global__ void ghm_zero_ws(float* ws) {
    if (threadIdx.x < 48) ws[threadIdx.x] = 0.0f;
}

__global__ __launch_bounds__(256) void ghm_main(const float* __restrict__ logits,
                                                const int*   __restrict__ targets,
                                                float*       __restrict__ ws,
                                                int n) {
    __shared__ float sHist[48];
    __shared__ int   sBin[256];
    __shared__ float sCe[256];

    const int tid  = threadIdx.x;
    const int lane = tid & 31;
    const int wib  = tid >> 5;               // wave in block (8 waves)
    if (tid < 48) sHist[tid] = 0.0f;

    const long long waveId  = (long long)blockIdx.x * 8 + wib;
    const int       rowBase = (int)(waveId * 32);

    const int rquot = lane >> 2;             // which of 8 rows this lane serves
    const int sub   = lane & 3;              // 32-float chunk within the row

    // Each wave covers 32 rows in 4 batches of 8 rows; 4 lanes per row,
    // 8 independent global_load_b128 per lane per batch (4 KB in flight/wave).
    for (int batch = 0; batch < 4; ++batch) {
        int myRow  = rowBase + batch * 8 + rquot;
        bool valid = myRow < n;
        int  rowC  = valid ? myRow : (n - 1);

        const float* rp = logits + (size_t)rowC * 128u;
        const f4*    cp = (const f4*)rp + sub * 8;

        f4 d[8];
        #pragma unroll
        for (int j = 0; j < 8; ++j)
            d[j] = __builtin_nontemporal_load(cp + j);   // streaming: read once

        int   t  = targets[rowC];
        float xt = rp[t];                                 // gather; hits cache

        // max over this lane's 32 elements, then 2-step group-of-4 reduction
        float m = fmaxf(fmaxf(d[0].x, d[0].y), fmaxf(d[0].z, d[0].w));
        #pragma unroll
        for (int j = 1; j < 8; ++j)
            m = fmaxf(m, fmaxf(fmaxf(d[j].x, d[j].y), fmaxf(d[j].z, d[j].w)));
        m = fmaxf(m, __shfl_xor(m, 1, 32));
        m = fmaxf(m, __shfl_xor(m, 2, 32));

        float s = 0.0f;
        #pragma unroll
        for (int j = 0; j < 8; ++j)
            s += __expf(d[j].x - m) + __expf(d[j].y - m) +
                 __expf(d[j].z - m) + __expf(d[j].w - m);
        s += __shfl_xor(s, 1, 32);
        s += __shfl_xor(s, 2, 32);

        float ce = (m - xt) + __logf(s);                  // -log_softmax[target]
        float g  = 2.0f * (1.0f - __expf(xt - m) / s);    // sum|p - onehot|

        // searchsorted(edges, g, 'left') - 1, clipped
        int idx = 0;
        #pragma unroll
        for (int j = 0; j <= GHM_BINS; ++j) {
            float edge = (j == GHM_BINS) ? (1.0f + 1e-6f) : (float)j * 0.1f;
            idx += (edge < g) ? 1 : 0;
        }
        int bin = idx - 1;
        bin = bin < 0 ? 0 : (bin > GHM_BINS - 1 ? GHM_BINS - 1 : bin);

        if (sub == 0) {                      // one writer per row -> staging
            int slot = wib * 32 + batch * 8 + rquot;
            sBin[slot] = valid ? bin : 15;   // bins 10..15 = ignored spill rows
            sCe [slot] = ce;
        }
    }
    __syncthreads();

    const int*   wb = &sBin[wib * 32];
    const float* wc = &sCe [wib * 32];
    const int  mrow = lane & 15;
    const bool hi   = lane >= 16;

    // A (16x32 f16) = one-hot(bin): A[m][k] = (bin_k == m). Contiguous k order:
    // lanes 0-15 hold K in {0..15} (h0..h15); lanes 16-31: {16..31}
    v16h a;
    #pragma unroll
    for (int tq = 0; tq < 8; ++tq) {
        int k0 = (hi ? 16 : 0) + tq;
        int k1 = k0 + 8;
        a[tq]     = (wb[k0] == mrow) ? (_Float16)1.0f : (_Float16)0.0f;
        a[8 + tq] = (wb[k1] == mrow) ? (_Float16)1.0f : (_Float16)0.0f;
    }

    // B (32x16 f16): B[k][n] = feature n of sample k; n=0 -> 1.0 (count),
    // n=1 -> f16-hi(ce), n=2 -> f16 residual. Lane n holds column n,
    // rows 0..15 (lanes 0-15) / rows 16..31 (lanes 16-31).
    v16h b;
    #pragma unroll
    for (int q = 0; q < 16; ++q) {
        int k = (hi ? 16 : 0) + q;
        float    ce = wc[k];
        _Float16 ch = (_Float16)ce;
        float    cl = ce - (float)ch;
        _Float16 val = (_Float16)0.0f;
        if      (mrow == 0) val = (_Float16)1.0f;
        else if (mrow == 1) val = ch;
        else if (mrow == 2) val = (_Float16)cl;
        b[q] = val;
    }

    // segment-sum via matrix core: D[bin][feature] += one-hot x features
    v8f c = {};
    c = __builtin_amdgcn_wmma_f32_16x16x32_f16(false, a, false, b,
                                               (short)0, c, false, false);

    // D layout: VGPR r -> M=r (lanes 0-15), M=r+8 (lanes 16-31); N = lane%16
    const int feat = lane & 15;
    if (feat < 3) {
        #pragma unroll
        for (int rr = 0; rr < 8; ++rr) {
            int bin = rr + (hi ? 8 : 0);
            if (bin < GHM_BINS) atomicAdd(&sHist[feat * 16 + bin], c[rr]);
        }
    }
    __syncthreads();
    if (tid < 48) atomicAdd(&ws[tid], sHist[tid]);
}

__global__ void ghm_final(const float* __restrict__ ws,
                          const float* __restrict__ acc_sum,
                          float* __restrict__ out) {
    int b = threadIdx.x;
    float num = 0.0f, den = 0.0f;
    if (b < GHM_BINS) {
        float cnt    = ws[b];
        float ceSum  = ws[16 + b] + ws[32 + b];
        float accNew = GHM_MOM * acc_sum[b] + (1.0f - GHM_MOM) * cnt;
        float inv    = 1.0f / (accNew + 2e-7f);   // 1/(density + 1e-7), density = acc+1e-7
        num = ceSum * inv;
        den = cnt   * inv;
    }
    #pragma unroll
    for (int off = 16; off > 0; off >>= 1) {
        num += __shfl_xor(num, off, 32);
        den += __shfl_xor(den, off, 32);
    }
    if (b == 0) out[0] = num / den;   // == mean(ce * ghm_w) * LOSS_WEIGHT
}

extern "C" void kernel_launch(void* const* d_in, const int* in_sizes, int n_in,
                              void* d_out, int out_size, void* d_ws, size_t ws_size,
                              hipStream_t stream) {
    const float* logits  = (const float*)d_in[0];
    const float* acc_sum = (const float*)d_in[1];
    const int*   targets = (const int*)d_in[2];
    float* out = (float*)d_out;
    float* ws  = (float*)d_ws;

    int n = in_sizes[2];                 // number of rows (B); C fixed at 128

    ghm_zero_ws<<<1, 64, 0, stream>>>(ws);

    int waves  = (n + 31) / 32;
    int blocks = (waves + 7) / 8;        // 8 waves / 256-thread block
    ghm_main<<<blocks, 256, 0, stream>>>(logits, targets, ws, n);

    ghm_final<<<1, 32, 0, stream>>>(ws, acc_sum, out);
}
